// DepthAwareGATv2_15522011808337
// MI455X (gfx1250) — matrix operand order repaired and checked by reading, not compiled, not run on hardware
//
#include <hip/hip_runtime.h>
#include <math.h>

typedef __bf16 bf16_t;
typedef __attribute__((ext_vector_type(16))) __bf16 v16bf;
typedef __attribute__((ext_vector_type(8)))  __bf16 v8bf;
typedef __attribute__((ext_vector_type(8)))  float  v8f;

#define D128 128
#define NH   8
#define HC   16

// ---------------------------------------------------------------- utilities
__device__ __forceinline__ void atomicMaxF(float* addr, float v) {
  if (v >= 0.f) atomicMax((int*)addr, __float_as_int(v));
  else          atomicMin((unsigned int*)addr, (unsigned int)__float_as_int(v));
}

__global__ void k_f2bf(const float* __restrict__ in, bf16_t* __restrict__ out, long long n) {
  long long i = (long long)blockIdx.x * blockDim.x + threadIdx.x;
  if (i < n) out[i] = (bf16_t)in[i];
}

__global__ void k_fill(float* __restrict__ p, float v, long long n) {
  long long i = (long long)blockIdx.x * blockDim.x + threadIdx.x;
  if (i < n) p[i] = v;
}

// ---------------------------------------------------------------- B-panel packer
// Pack f32 weight B[K,Ncol] into WMMA B-fragment order (zero-fill cols >= Ncol):
//   Bp[ ((tn*KS + ks)*32 + lane)*16 + e ] = B[(ks*32 + (lane>>4)*16 + e)*ldb + tn*16 + (lane&15)]
__global__ void k_pack_b(const float* __restrict__ B, int ldb, int Ncol, int K,
                         bf16_t* __restrict__ Bp, int tilesN)
{
  int KS = K >> 5;
  int total = tilesN * KS * 512;
  int idx = blockIdx.x * blockDim.x + threadIdx.x;
  if (idx >= total) return;
  int e    = idx & 15;
  int lane = (idx >> 4) & 31;
  int blk  = idx >> 9;
  int tn   = blk / KS;
  int ks   = blk - tn * KS;
  int col  = tn * 16 + (lane & 15);
  int kk   = ks * 32 + (lane >> 4) * 16 + e;
  Bp[idx] = (col < Ncol) ? (bf16_t)B[(size_t)kk * ldb + col] : (bf16_t)0.f;
}

// ---------------------------------------------------------------- bf16 WMMA GEMM
// C[M,Ncol] = A[M,K](bf16,lda) @ Bpacked + bias. One wave per 16x16 tile.
// K compile-time (multiple of 32); M multiple of 16; epilogue statically selected.
template <int K, bool HAS_BIAS, bool HAS_CF, bool HAS_CB>
__global__ void k_gemm_wmma_bf16(const bf16_t* __restrict__ A, int lda,
                                 const bf16_t* __restrict__ Bp,
                                 const float* __restrict__ bias,
                                 float*  __restrict__ Cf, int ldc,
                                 bf16_t* __restrict__ Cb, int ldcb,
                                 int M, int Ncol, int tilesN)
{
  constexpr int KS = K / 32;
  int gw   = (int)(((long long)blockIdx.x * blockDim.x + threadIdx.x) >> 5);
  int lane = threadIdx.x & 31;
  int tilesM = M >> 4;
  if (gw >= tilesM * tilesN) return;
  int tm = gw / tilesN;
  int tn = gw - tm * tilesN;
  int half = lane >> 4;
  int r    = lane & 15;

  const bf16_t* ap = A + (size_t)(tm * 16 + r) * lda + half * 16;
  const bf16_t* bp = Bp + (size_t)tn * KS * 512 + lane * 16;

  v8f acc = {};
#pragma unroll
  for (int ks = 0; ks < KS; ++ks) {
    v8bf a0 = *(const v8bf*)(ap);        // K = ks*32 + 16*half + [0..8)
    v8bf a1 = *(const v8bf*)(ap + 8);    // K = ks*32 + 16*half + 8 + [0..8)
    v8bf b0 = *(const v8bf*)(bp);
    v8bf b1 = *(const v8bf*)(bp + 8);
    v16bf a, b;
#pragma unroll
    for (int e = 0; e < 8; ++e) {
      a[e] = a0[e]; a[e + 8] = a1[e];
      b[e] = b0[e]; b[e + 8] = b1[e];
    }
    acc = __builtin_amdgcn_wmma_f32_16x16x32_bf16(false, a, false, b,
                                                  (short)0, acc, false, false);
    ap += 32;
    bp += 512;
  }

  int col = tn * 16 + r;
  if (col >= Ncol) return;
  float bv = HAS_BIAS ? bias[col] : 0.f;
#pragma unroll
  for (int v = 0; v < 8; ++v) {           // C/D: VGPR v -> row = 8*half + v
    int row = tm * 16 + half * 8 + v;
    float val = acc[v] + bv;
    if (HAS_CF) Cf[(size_t)row * ldc  + col] = val;
    if (HAS_CB) Cb[(size_t)row * ldcb + col] = (bf16_t)val;
  }
}

// ---------------------------------------------------------------- edge feature MLP
__global__ void k_edge_poly_mlp(const float* __restrict__ x,
                                const long long* __restrict__ ei, long long E,
                                const float* __restrict__ w1, const float* __restrict__ b1,
                                const float* __restrict__ w2, const float* __restrict__ b2,
                                float* __restrict__ edge_attr)
{
  long long e = ((long long)blockIdx.x * blockDim.x + threadIdx.x) >> 5;
  int lane = threadIdx.x & 31;
  if (e >= E) return;
  long long s = ei[e], d = ei[E + e];
  float4 a = ((const float4*)(x + s * D128))[lane];
  float4 b = ((const float4*)(x + d * D128))[lane];
  float dot = a.x * b.x + a.y * b.y + a.z * b.z + a.w * b.w;
#pragma unroll
  for (int off = 16; off; off >>= 1) dot += __shfl_xor(dot, off, 32);
  float sN = dot * (1.f / 128.f);
  float p0 = sN, p1 = sN * sN, p2 = p1 * sN, p3 = p2 * sN;
  float hid = 0.f;
  if (lane < 16) {
    hid = b1[lane] + p0 * w1[0 * 16 + lane] + p1 * w1[1 * 16 + lane]
        + p2 * w1[2 * 16 + lane] + p3 * w1[3 * 16 + lane];
    hid = fmaxf(hid, 0.f);
  }
  float out = (lane < NH) ? b2[lane] : 0.f;
#pragma unroll
  for (int j = 0; j < 16; ++j) {
    float hj = __shfl(hid, j, 32);
    if (lane < NH) out += hj * w2[j * NH + lane];
  }
  if (lane < NH) edge_attr[e * NH + lane] = out;
}

// fold: F[j][h] = sum_c edge_w[j, h*16+c] * att_edge[h,c]   (8x8)
__global__ void k_fold(const float* __restrict__ edge_w, const float* __restrict__ att_edge,
                       float* __restrict__ F)
{
  int t = threadIdx.x;
  if (t >= 64) return;
  int j = t >> 3, h = t & 7;
  float s = 0.f;
#pragma unroll
  for (int c = 0; c < HC; ++c) s += edge_w[j * D128 + h * HC + c] * att_edge[h * HC + c];
  F[j * NH + h] = s;
}

// a_e for both layers: [E,8] @ F(8x8)
__global__ void k_edge_ae(const float* __restrict__ ea,
                          const float* __restrict__ F1, const float* __restrict__ F2,
                          float* __restrict__ ae1, float* __restrict__ ae2, long long E)
{
  long long e = (long long)blockIdx.x * blockDim.x + threadIdx.x;
  if (e >= E) return;
  float v[NH];
#pragma unroll
  for (int j = 0; j < NH; ++j) v[j] = ea[e * NH + j];
#pragma unroll
  for (int h = 0; h < NH; ++h) {
    float s1 = 0.f, s2 = 0.f;
#pragma unroll
    for (int j = 0; j < NH; ++j) { s1 += v[j] * F1[j * NH + h]; s2 += v[j] * F2[j * NH + h]; }
    ae1[e * NH + h] = s1;
    ae2[e * NH + h] = s2;
  }
}

// per (node, head): a_src/a_dst from xs
__global__ void k_node_att(const float* __restrict__ xs,
                           const float* __restrict__ att_src, const float* __restrict__ att_dst,
                           float* __restrict__ asrc, float* __restrict__ adst, int Nn)
{
  int i = blockIdx.x * blockDim.x + threadIdx.x;
  if (i >= Nn * NH) return;
  int n = i >> 3, h = i & 7;
  const float* xr = xs + (size_t)n * D128 + h * HC;
  float s1 = 0.f, s2 = 0.f;
#pragma unroll
  for (int c = 0; c < HC; ++c) { float v = xr[c]; s1 += v * att_src[h * HC + c]; s2 += v * att_dst[h * HC + c]; }
  asrc[i] = s1;
  adst[i] = s2;
}

// per (edge, head): alpha = leaky_relu(...), segment max into m[dst]
__global__ void k_edge_alpha_max(const long long* __restrict__ ei,
                                 const float* __restrict__ asrc, const float* __restrict__ adst,
                                 const float* __restrict__ ae,
                                 float* __restrict__ alpha, float* __restrict__ m, long long E)
{
  long long i = (long long)blockIdx.x * blockDim.x + threadIdx.x;
  if (i >= E * NH) return;
  long long e = i >> 3; int h = (int)(i & 7);
  long long s = ei[e], d = ei[E + e];
  float a = asrc[s * NH + h] + adst[d * NH + h] + ae[i];
  a = a > 0.f ? a : 0.2f * a;
  alpha[i] = a;
  atomicMaxF(&m[d * NH + h], a);
}

// per (edge, head): a = exp(alpha - m[dst]); segment sum
__global__ void k_edge_expsum(const long long* __restrict__ ei,
                              float* __restrict__ alpha, const float* __restrict__ m,
                              float* __restrict__ ssum, long long E)
{
  long long i = (long long)blockIdx.x * blockDim.x + threadIdx.x;
  if (i >= E * NH) return;
  long long e = i >> 3; int h = (int)(i & 7);
  long long d = ei[E + e];
  float ex = expf(alpha[i] - m[d * NH + h]);
  alpha[i] = ex;
  atomicAdd(&ssum[d * NH + h], ex);
}

// per (edge, 4-channel group): scatter-add xs[src]*w into out[dst]
__global__ void k_edge_msg(const long long* __restrict__ ei,
                           const float* __restrict__ xs,
                           const float* __restrict__ aexp, const float* __restrict__ ssum,
                           float* __restrict__ out, long long E)
{
  long long i = (long long)blockIdx.x * blockDim.x + threadIdx.x;
  if (i >= E * 32) return;
  long long e = i >> 5; int g = (int)(i & 31); int h = g >> 2;
  long long s = ei[e], d = ei[E + e];
  float w = aexp[e * NH + h] / (ssum[d * NH + h] + 1e-16f);
  float4 v = ((const float4*)(xs + s * D128))[g];
  float* o = out + d * D128 + g * 4;
  atomicAdd(o + 0, v.x * w);
  atomicAdd(o + 1, v.y * w);
  atomicAdd(o + 2, v.z * w);
  atomicAdd(o + 3, v.w * w);
}

// wave per node: h = elu(conv + bias) + resid; LayerNorm; write f32 + bf16(into hcat)
template <bool HAS_HOUT>
__global__ void k_post_ln(const float* __restrict__ conv, const float* __restrict__ bias,
                          const float* __restrict__ resid,
                          const float* __restrict__ g, const float* __restrict__ b,
                          float* __restrict__ hout, bf16_t* __restrict__ hcat, int catOff, int Nn)
{
  int node = (int)(((long long)blockIdx.x * blockDim.x + threadIdx.x) >> 5);
  int lane = threadIdx.x & 31;
  if (node >= Nn) return;
  float4 c  = ((const float4*)(conv  + (size_t)node * D128))[lane];
  float4 bs = ((const float4*)bias)[lane];
  float4 rr = ((const float4*)(resid + (size_t)node * D128))[lane];
  float t[4] = { c.x + bs.x, c.y + bs.y, c.z + bs.z, c.w + bs.w };
  float rv[4] = { rr.x, rr.y, rr.z, rr.w };
  float sum = 0.f, sq = 0.f;
#pragma unroll
  for (int k = 0; k < 4; ++k) {
    float v = t[k] > 0.f ? t[k] : (expf(t[k]) - 1.f);   // ELU
    v += rv[k];
    t[k] = v; sum += v; sq += v * v;
  }
#pragma unroll
  for (int off = 16; off; off >>= 1) { sum += __shfl_xor(sum, off, 32); sq += __shfl_xor(sq, off, 32); }
  float mu = sum * (1.f / 128.f);
  float var = sq * (1.f / 128.f) - mu * mu;
  float rstd = rsqrtf(var + 1e-5f);
  float4 g4 = ((const float4*)g)[lane];
  float4 b4 = ((const float4*)b)[lane];
  float gg[4] = { g4.x, g4.y, g4.z, g4.w };
  float bb[4] = { b4.x, b4.y, b4.z, b4.w };
#pragma unroll
  for (int k = 0; k < 4; ++k) {
    float v = (t[k] - mu) * rstd * gg[k] + bb[k];
    if (HAS_HOUT) hout[(size_t)node * D128 + lane * 4 + k] = v;
    hcat[(size_t)node * 384 + catOff + lane * 4 + k] = (bf16_t)v;
  }
}

// wave per node: in-place log_softmax over C=40 columns
__global__ void k_logsoftmax(float* __restrict__ out, int Nn, int C)
{
  int node = (int)(((long long)blockIdx.x * blockDim.x + threadIdx.x) >> 5);
  int lane = threadIdx.x & 31;
  if (node >= Nn) return;
  float* row = out + (size_t)node * C;
  float v0 = lane      < C ? row[lane]      : -3.4e38f;
  float v1 = lane + 32 < C ? row[lane + 32] : -3.4e38f;
  float mx = fmaxf(v0, v1);
#pragma unroll
  for (int off = 16; off; off >>= 1) mx = fmaxf(mx, __shfl_xor(mx, off, 32));
  float s = (lane < C ? expf(v0 - mx) : 0.f) + (lane + 32 < C ? expf(v1 - mx) : 0.f);
#pragma unroll
  for (int off = 16; off; off >>= 1) s += __shfl_xor(s, off, 32);
  float lse = mx + logf(s);
  if (lane      < C) row[lane]      = v0 - lse;
  if (lane + 32 < C) row[lane + 32] = v1 - lse;
}

// ---------------------------------------------------------------- launcher
extern "C" void kernel_launch(void* const* d_in, const int* in_sizes, int n_in,
                              void* d_out, int out_size, void* d_ws, size_t ws_size,
                              hipStream_t stream)
{
  const float*     x       = (const float*)d_in[0];
  const long long* ei      = (const long long*)d_in[1];
  const float* ee_w1 = (const float*)d_in[2];
  const float* ee_b1 = (const float*)d_in[3];
  const float* ee_w2 = (const float*)d_in[4];
  const float* ee_b2 = (const float*)d_in[5];
  const float* in_w  = (const float*)d_in[6];
  const float* in_b  = (const float*)d_in[7];
  const float* c1_lin_w   = (const float*)d_in[8];
  const float* c1_edge_w  = (const float*)d_in[9];
  const float* c1_att_src = (const float*)d_in[10];
  const float* c1_att_dst = (const float*)d_in[11];
  const float* c1_att_edge= (const float*)d_in[12];
  const float* c1_bias    = (const float*)d_in[13];
  const float* c2_lin_w   = (const float*)d_in[14];
  const float* c2_edge_w  = (const float*)d_in[15];
  const float* c2_att_src = (const float*)d_in[16];
  const float* c2_att_dst = (const float*)d_in[17];
  const float* c2_att_edge= (const float*)d_in[18];
  const float* c2_bias    = (const float*)d_in[19];
  const float* n1_g = (const float*)d_in[20];
  const float* n1_b = (const float*)d_in[21];
  const float* n2_g = (const float*)d_in[22];
  const float* n2_b = (const float*)d_in[23];
  const float* jk_w = (const float*)d_in[24];
  const float* jk_b = (const float*)d_in[25];
  const float* cls_w = (const float*)d_in[26];
  const float* cls_b = (const float*)d_in[27];

  const int       Nn = in_sizes[0] / D128;       // 50000
  const long long E  = in_sizes[1] / 2;          // 800000
  const int       NCLS = 40;
  const int       TN_D = D128 / 16;              // 8
  const int       TN_C = (NCLS + 15) / 16;       // 3

  // ---- workspace carve-up (256B aligned)
  char* base = (char*)d_ws;
  size_t off = 0;
  auto carve = [&](size_t bytes) -> char* {
    char* p = base + off;
    off += (bytes + 255) & ~(size_t)255;
    return p;
  };
  bf16_t* x_bf    = (bf16_t*)carve((size_t)Nn * D128 * 2);
  bf16_t* Bp_in   = (bf16_t*)carve((size_t)TN_D * 4  * 512 * 2);
  bf16_t* Bp_l1   = (bf16_t*)carve((size_t)TN_D * 4  * 512 * 2);
  bf16_t* Bp_l2   = (bf16_t*)carve((size_t)TN_D * 4  * 512 * 2);
  bf16_t* Bp_jk   = (bf16_t*)carve((size_t)TN_D * 12 * 512 * 2);
  bf16_t* Bp_cls  = (bf16_t*)carve((size_t)TN_C * 4  * 512 * 2);
  float*  F1      = (float*)carve(64 * 4);
  float*  F2      = (float*)carve(64 * 4);
  float*  edge_attr = (float*)carve((size_t)E * NH * 4);
  float*  ae1     = (float*)carve((size_t)E * NH * 4);
  float*  ae2     = (float*)carve((size_t)E * NH * 4);
  float*  eatt    = (float*)carve((size_t)E * NH * 4);   // alpha, then exp(alpha)
  float*  h0      = (float*)carve((size_t)Nn * D128 * 4);
  float*  h1      = (float*)carve((size_t)Nn * D128 * 4);
  float*  xs      = (float*)carve((size_t)Nn * D128 * 4);
  float*  conv    = (float*)carve((size_t)Nn * D128 * 4);
  float*  asrc    = (float*)carve((size_t)Nn * NH * 4);
  float*  adst    = (float*)carve((size_t)Nn * NH * 4);
  float*  mbuf    = (float*)carve((size_t)Nn * NH * 4);
  float*  sbuf    = (float*)carve((size_t)Nn * NH * 4);
  bf16_t* hcat    = (bf16_t*)carve((size_t)Nn * 384 * 2);
  bf16_t* hjk_bf  = (bf16_t*)carve((size_t)Nn * D128 * 2);
  (void)ws_size; (void)n_in; (void)out_size;

  const int BS = 256;
  auto blocksFor = [](long long n, int bs) { return (unsigned)((n + bs - 1) / bs); };

  // ---- precompute: bf16 activations, packed weight panels, folded 8x8 matrices
  k_f2bf<<<blocksFor((long long)Nn * D128, BS), BS, 0, stream>>>(x, x_bf, (long long)Nn * D128);
  k_pack_b<<<blocksFor((long long)TN_D * 4  * 512, BS), BS, 0, stream>>>(in_w,     D128, D128, D128, Bp_in,  TN_D);
  k_pack_b<<<blocksFor((long long)TN_D * 4  * 512, BS), BS, 0, stream>>>(c1_lin_w, D128, D128, D128, Bp_l1,  TN_D);
  k_pack_b<<<blocksFor((long long)TN_D * 4  * 512, BS), BS, 0, stream>>>(c2_lin_w, D128, D128, D128, Bp_l2,  TN_D);
  k_pack_b<<<blocksFor((long long)TN_D * 12 * 512, BS), BS, 0, stream>>>(jk_w,     D128, D128, 384,  Bp_jk,  TN_D);
  k_pack_b<<<blocksFor((long long)TN_C * 4  * 512, BS), BS, 0, stream>>>(cls_w,    NCLS, NCLS, D128, Bp_cls, TN_C);
  k_fold<<<1, 64, 0, stream>>>(c1_edge_w, c1_att_edge, F1);
  k_fold<<<1, 64, 0, stream>>>(c2_edge_w, c2_att_edge, F2);

  // ---- edge features + tiny MLP + folded a_e for both layers
  k_edge_poly_mlp<<<blocksFor(E * 32, BS), BS, 0, stream>>>(x, ei, E, ee_w1, ee_b1, ee_w2, ee_b2, edge_attr);
  k_edge_ae<<<blocksFor(E, BS), BS, 0, stream>>>(edge_attr, F1, F2, ae1, ae2, E);

  // ---- h0 = x @ in_w + in_b   (f32 h0 + bf16 into hcat[:,0:128])
  {
    long long waves = (long long)(Nn / 16) * TN_D;
    k_gemm_wmma_bf16<128, true, true, true><<<blocksFor(waves * 32, BS), BS, 0, stream>>>(
        x_bf, D128, Bp_in, in_b, h0, D128, hcat, 384, Nn, D128, TN_D);
  }

  // ---- two GAT layers
  for (int layer = 0; layer < 2; ++layer) {
    const bf16_t* Ain  = layer == 0 ? x_bf : (hcat + 128);
    int           lda  = layer == 0 ? D128 : 384;
    const bf16_t* Bp   = layer == 0 ? Bp_l1 : Bp_l2;
    const float*  atS  = layer == 0 ? c1_att_src : c2_att_src;
    const float*  atD  = layer == 0 ? c1_att_dst : c2_att_dst;
    const float*  ae   = layer == 0 ? ae1 : ae2;
    const float*  cb   = layer == 0 ? c1_bias : c2_bias;
    const float*  res  = layer == 0 ? h0 : h1;
    const float*  lg   = layer == 0 ? n1_g : n2_g;
    const float*  lb   = layer == 0 ? n1_b : n2_b;
    int           cOff = layer == 0 ? 128 : 256;

    long long waves = (long long)(Nn / 16) * TN_D;
    k_gemm_wmma_bf16<128, false, true, false><<<blocksFor(waves * 32, BS), BS, 0, stream>>>(
        Ain, lda, Bp, nullptr, xs, D128, nullptr, 0, Nn, D128, TN_D);

    k_node_att<<<blocksFor((long long)Nn * NH, BS), BS, 0, stream>>>(xs, atS, atD, asrc, adst, Nn);

    k_fill<<<blocksFor((long long)Nn * NH, BS), BS, 0, stream>>>(mbuf, -1e30f, (long long)Nn * NH);
    k_fill<<<blocksFor((long long)Nn * NH, BS), BS, 0, stream>>>(sbuf, 0.f, (long long)Nn * NH);
    k_fill<<<blocksFor((long long)Nn * D128, BS), BS, 0, stream>>>(conv, 0.f, (long long)Nn * D128);

    k_edge_alpha_max<<<blocksFor(E * NH, BS), BS, 0, stream>>>(ei, asrc, adst, ae, eatt, mbuf, E);
    k_edge_expsum<<<blocksFor(E * NH, BS), BS, 0, stream>>>(ei, eatt, mbuf, sbuf, E);
    k_edge_msg<<<blocksFor(E * 32, BS), BS, 0, stream>>>(ei, xs, eatt, sbuf, conv, E);

    if (layer == 0)
      k_post_ln<true><<<blocksFor((long long)Nn * 32, BS), BS, 0, stream>>>(
          conv, cb, res, lg, lb, h1, hcat, cOff, Nn);
    else
      k_post_ln<false><<<blocksFor((long long)Nn * 32, BS), BS, 0, stream>>>(
          conv, cb, res, lg, lb, nullptr, hcat, cOff, Nn);
  }

  // ---- JK: [h0|h1|h2] @ jk_w + jk_b  -> bf16 hjk
  {
    long long waves = (long long)(Nn / 16) * TN_D;
    k_gemm_wmma_bf16<384, true, false, true><<<blocksFor(waves * 32, BS), BS, 0, stream>>>(
        hcat, 384, Bp_jk, jk_b, nullptr, 0, hjk_bf, D128, Nn, D128, TN_D);
  }

  // ---- classifier: hjk @ cls_w + cls_b -> logits (f32, d_out), then log_softmax
  {
    long long waves = (long long)(Nn / 16) * TN_C;
    k_gemm_wmma_bf16<128, true, true, false><<<blocksFor(waves * 32, BS), BS, 0, stream>>>(
        hjk_bf, D128, Bp_cls, cls_b, (float*)d_out, NCLS, nullptr, 0, Nn, NCLS, TN_C);
  }
  k_logsoftmax<<<blocksFor((long long)Nn * 32, BS), BS, 0, stream>>>((float*)d_out, Nn, NCLS);
}
